// PanoramicEncoder_13829794693353
// MI455X (gfx1250) — matrix operand order repaired and checked by reading, not compiled, not run on hardware
//
#include <hip/hip_runtime.h>
#include <hip/hip_bf16.h>

// ---------------------------------------------------------------------------
// BERT-base forward for MI455X (gfx1250, wave32, WMMA bf16 path)
// NL=12 NH=12 H=768 FF=3072 V=30522 P=512 B=4 L=512 S=8 DH=64
// ---------------------------------------------------------------------------

#define NLAYER 12
#define NHEAD  12
#define HDIM   768
#define FFDIM  3072
#define DHEAD  64
#define BATCH  4
#define SEQ    512
#define NSEG   8
#define NROW   (BATCH*SEQ)   // 2048

typedef __attribute__((ext_vector_type(16))) __bf16 v16bf;
typedef __attribute__((ext_vector_type(8)))  float  v8f;

union Frag {
    uint4 q[2];
    v16bf v;
};

__device__ __forceinline__ unsigned f2bf(float f) {
    union { float f; unsigned u; } x; x.f = f;
    unsigned u = x.u;
    // round-to-nearest-even bf16
    return (u + 0x7fffu + ((u >> 16) & 1u)) >> 16;
}

__device__ __forceinline__ float gelu_exact(float v) {
    return 0.5f * v * (1.0f + erff(v * 0.70710678118654752f));
}

// ---------------------------------------------------------------------------
// GEMM: C[M x N] = epilogue(A_bf16[M x K] @ W_f32[K x N] + bias[N])
// MODE 0: fp32 out   MODE 1: bf16 out   MODE 2: gelu -> bf16 out
// Block: 256 threads (8 waves). Tile: BM=128, BN=128, BK=32.
// Double-buffered LDS, register-staged global loads, 1 barrier per K-step.
// ---------------------------------------------------------------------------
#define BM 128
#define BN 128
#define BK 32
#define LDT 40   // LDS row stride in halves (32 + 8 pad, keeps 16B alignment)

template<int MODE>
__global__ __launch_bounds__(256) void gemm_bf16_kernel(
    const unsigned short* __restrict__ A,   // M x K, bf16
    const float* __restrict__ W,            // K x N, fp32 (converted on the fly)
    const float* __restrict__ bias,         // N
    void* __restrict__ out,                 // M x N
    int M, int N, int K)
{
    __shared__ __align__(16) unsigned short sA[2][BM * LDT]; // A tile (M x K)
    __shared__ __align__(16) unsigned short sB[2][BN * LDT]; // W tile transposed (N x K)

    const int tid  = threadIdx.x;
    const int wave = tid >> 5;
    const int lane = tid & 31;
    const int bm = blockIdx.y * BM;
    const int bn = blockIdx.x * BN;
    const int m0 = wave * 16;

    v8f acc[8];
#pragma unroll
    for (int i = 0; i < 8; i++)
#pragma unroll
        for (int j = 0; j < 8; j++) acc[i][j] = 0.0f;

    // A-tile load map: thread t -> row t/2, halves [(t&1)*16, +16)
    const int ar = tid >> 1;
    const int ac = (tid & 1) * 16;
    // W-tile load map: thread t -> k-pair (t&15)*2, 8 columns at (t>>4)*8
    const int wk = (tid & 15) * 2;
    const int wn = (tid >> 4) * 8;

    uint4  aR0, aR1;
    float4 wR0, wR1, wR2, wR3;

#define GEMM_LOADG(K0)                                                          \
    do {                                                                        \
        const uint4* _as = (const uint4*)(A + (size_t)(bm + ar) * K + (K0) + ac);\
        aR0 = _as[0]; aR1 = _as[1];                                             \
        const float4* _w0 = (const float4*)(W + (size_t)((K0) + wk) * N + bn + wn);   \
        const float4* _w1 = (const float4*)(W + (size_t)((K0) + wk + 1) * N + bn + wn);\
        wR0 = _w0[0]; wR1 = _w0[1]; wR2 = _w1[0]; wR3 = _w1[1];                 \
    } while (0)

#define GEMM_STORELDS(BUF)                                                      \
    do {                                                                        \
        uint4* _ad = (uint4*)(sA[BUF] + ar * LDT + ac);                         \
        _ad[0] = aR0; _ad[1] = aR1;                                             \
        const float* _f0 = (const float*)&wR0;  /* k = wk,   n = wn..wn+7 */    \
        const float* _f1 = (const float*)&wR2;  /* k = wk+1, n = wn..wn+7 */    \
        _Pragma("unroll")                                                       \
        for (int _i = 0; _i < 8; _i++) {                                        \
            unsigned _pk = f2bf(_f0[_i]) | (f2bf(_f1[_i]) << 16);               \
            *(unsigned*)(sB[BUF] + (wn + _i) * LDT + wk) = _pk;                 \
        }                                                                       \
    } while (0)

    GEMM_LOADG(0);
    GEMM_STORELDS(0);
    __syncthreads();

    int buf = 0;
    const int ks = (lane & 16) ? 16 : 0;
    for (int k0 = 0; k0 < K; k0 += BK) {
        const bool has_next = (k0 + BK) < K;
        if (has_next) GEMM_LOADG(k0 + BK);

        // A fragment: lane<16 -> K {0..15}; lane>=16 -> K {16..31}
        Frag a;
        const unsigned short* ap = sA[buf] + (m0 + (lane & 15)) * LDT + ks;
        a.q[0] = *(const uint4*)ap;
        a.q[1] = *(const uint4*)(ap + 8);

#pragma unroll
        for (int nt = 0; nt < 8; nt++) {
            // B fragment: lane -> column N, 16 contiguous K-halves from (lane>>4)*16
            Frag b;
            const unsigned short* bp = sB[buf] + (nt * 16 + (lane & 15)) * LDT + ((lane >> 4) * 16);
            b.q[0] = *(const uint4*)bp;
            b.q[1] = *(const uint4*)(bp + 8);
            acc[nt] = __builtin_amdgcn_wmma_f32_16x16x32_bf16(
                false, a.v, false, b.v, (short)0, acc[nt], false, false);
        }

        if (has_next) {
            GEMM_STORELDS(buf ^ 1);
            __syncthreads();
            buf ^= 1;
        }
    }
#undef GEMM_LOADG
#undef GEMM_STORELDS

    // epilogue. C layout: M = r + 8*(lane>>4), N = lane&15
#pragma unroll
    for (int nt = 0; nt < 8; nt++) {
        const int col = bn + nt * 16 + (lane & 15);
        const float bv = bias[col];
#pragma unroll
        for (int r = 0; r < 8; r++) {
            const int row = bm + m0 + r + 8 * (lane >> 4);
            float v = acc[nt][r] + bv;
            if (MODE == 2) v = gelu_exact(v);
            if (MODE == 0) ((float*)out)[(size_t)row * N + col] = v;
            else           ((unsigned short*)out)[(size_t)row * N + col] = (unsigned short)f2bf(v);
        }
    }
}

// ---------------------------------------------------------------------------
// mask metadata: valid_len[b], hist[b], seg[b][t]
// ---------------------------------------------------------------------------
__global__ void setup_mask_kernel(const int* __restrict__ am,
                                  const int* __restrict__ bos,
                                  int* __restrict__ seg,
                                  int* __restrict__ vlen,
                                  int* __restrict__ hist)
{
    int i = blockIdx.x * 256 + threadIdx.x;
    if (i < BATCH) {
        int s = 0;
        for (int t = 0; t < SEQ; t++) s += am[i * SEQ + t];
        vlen[i] = s;
        hist[i] = bos[i * NSEG];
    }
    if (i < NROW) {
        int b = i >> 9, t = i & 511;
        int c = 0;
        for (int s = 0; s < NSEG; s++) c += (bos[b * NSEG + s] <= t) ? 1 : 0;
        seg[i] = c - 1;
    }
}

// ---------------------------------------------------------------------------
// embedding gather + LayerNorm -> x (fp32) and xb (bf16). One block per row.
// ---------------------------------------------------------------------------
__global__ __launch_bounds__(256) void embed_ln_kernel(
    const float* __restrict__ we, const float* __restrict__ pe,
    const float* __restrict__ te, const float* __restrict__ g,
    const float* __restrict__ be, const int* __restrict__ ids,
    const int* __restrict__ tt, float* __restrict__ x,
    unsigned short* __restrict__ xb)
{
    const int row = blockIdx.x;            // b*512 + t
    const int t = row & 511;
    const int id = ids[row];
    const int ty = tt[row];
    __shared__ float red[256];

    float v[3];
    float sum = 0.0f;
#pragma unroll
    for (int j = 0; j < 3; j++) {
        int d = threadIdx.x + j * 256;
        v[j] = we[(size_t)id * HDIM + d] + pe[(size_t)t * HDIM + d] + te[(size_t)ty * HDIM + d];
        sum += v[j];
    }
    red[threadIdx.x] = sum; __syncthreads();
    for (int off = 128; off > 0; off >>= 1) {
        if (threadIdx.x < off) red[threadIdx.x] += red[threadIdx.x + off];
        __syncthreads();
    }
    float mean = red[0] * (1.0f / HDIM);
    __syncthreads();
    float s2 = 0.0f;
#pragma unroll
    for (int j = 0; j < 3; j++) { float d = v[j] - mean; s2 += d * d; }
    red[threadIdx.x] = s2; __syncthreads();
    for (int off = 128; off > 0; off >>= 1) {
        if (threadIdx.x < off) red[threadIdx.x] += red[threadIdx.x + off];
        __syncthreads();
    }
    float rstd = rsqrtf(red[0] * (1.0f / HDIM) + 1e-12f);
#pragma unroll
    for (int j = 0; j < 3; j++) {
        int d = threadIdx.x + j * 256;
        float y = (v[j] - mean) * rstd * g[d] + be[d];
        x[(size_t)row * HDIM + d] = y;
        xb[(size_t)row * HDIM + d] = (unsigned short)f2bf(y);
    }
}

// ---------------------------------------------------------------------------
// residual add + LayerNorm: x = LN(x + tmp) ; also refresh xb (bf16)
// ---------------------------------------------------------------------------
__global__ __launch_bounds__(256) void add_ln_kernel(
    float* __restrict__ x, const float* __restrict__ tmp,
    const float* __restrict__ g, const float* __restrict__ be,
    unsigned short* __restrict__ xb)
{
    const int row = blockIdx.x;
    __shared__ float red[256];
    float v[3];
    float sum = 0.0f;
#pragma unroll
    for (int j = 0; j < 3; j++) {
        int d = threadIdx.x + j * 256;
        v[j] = x[(size_t)row * HDIM + d] + tmp[(size_t)row * HDIM + d];
        sum += v[j];
    }
    red[threadIdx.x] = sum; __syncthreads();
    for (int off = 128; off > 0; off >>= 1) {
        if (threadIdx.x < off) red[threadIdx.x] += red[threadIdx.x + off];
        __syncthreads();
    }
    float mean = red[0] * (1.0f / HDIM);
    __syncthreads();
    float s2 = 0.0f;
#pragma unroll
    for (int j = 0; j < 3; j++) { float d = v[j] - mean; s2 += d * d; }
    red[threadIdx.x] = s2; __syncthreads();
    for (int off = 128; off > 0; off >>= 1) {
        if (threadIdx.x < off) red[threadIdx.x] += red[threadIdx.x + off];
        __syncthreads();
    }
    float rstd = rsqrtf(red[0] * (1.0f / HDIM) + 1e-12f);
#pragma unroll
    for (int j = 0; j < 3; j++) {
        int d = threadIdx.x + j * 256;
        float y = (v[j] - mean) * rstd * g[d] + be[d];
        x[(size_t)row * HDIM + d] = y;
        xb[(size_t)row * HDIM + d] = (unsigned short)f2bf(y);
    }
}

// ---------------------------------------------------------------------------
// V -> V^T per head: vt[b][h][d][t] = vb[b*512+t][h*64+d]
// ---------------------------------------------------------------------------
__global__ void transpose_v_kernel(const unsigned short* __restrict__ vb,
                                   unsigned short* __restrict__ vt)
{
    int i = blockIdx.x * 256 + threadIdx.x;
    if (i >= NROW * HDIM) return;
    int row = i / HDIM, col = i - row * HDIM;
    int b = row >> 9, t = row & 511;
    int h = col >> 6, d = col & 63;
    vt[(((size_t)(b * NHEAD + h)) * DHEAD + d) * SEQ + t] = vb[i];
}

// ---------------------------------------------------------------------------
// Fused attention scores + bias + softmax -> probs (bf16)
// One block (8 waves) per (b, h, 16-query tile). Q.K^T via WMMA bf16.
// Mask is computed branchlessly from pre-staged segment IDs.
// ---------------------------------------------------------------------------
__global__ __launch_bounds__(256) void attn_scores_softmax_kernel(
    const unsigned short* __restrict__ qb,   // [2048 x 768] bf16
    const unsigned short* __restrict__ kb,   // [2048 x 768] bf16
    unsigned short* __restrict__ probs,      // [B*NH*512 x 512] bf16
    const int* __restrict__ seg, const int* __restrict__ vlen,
    const int* __restrict__ hist)
{
    const int qt = blockIdx.x & 31;                 // query tile (L/16 = 32)
    const int h  = (blockIdx.x >> 5) % NHEAD;
    const int b  = blockIdx.x / (32 * NHEAD);
    const int tid = threadIdx.x, wave = tid >> 5, lane = tid & 31;

    __shared__ __align__(16) float s[16][SEQ];      // 32 KB score tile
    __shared__ float red[16][16];
    __shared__ int sq[16];                          // segment id per query row

    if (tid < 16) sq[tid] = seg[b * SEQ + qt * 16 + tid];

    // A fragments for the two K-steps of the 64-wide head dim
    const int ks = (lane & 16) ? 16 : 0;
    const int tqa = qt * 16 + (lane & 15);
    const unsigned short* qrow = qb + (size_t)(b * SEQ + tqa) * HDIM + h * DHEAD;
    Frag a0, a1;
    a0.q[0] = *(const uint4*)(qrow + ks);
    a0.q[1] = *(const uint4*)(qrow + ks + 8);
    a1.q[0] = *(const uint4*)(qrow + 32 + ks);
    a1.q[1] = *(const uint4*)(qrow + 32 + ks + 8);

    const int vl = vlen[b];
    const int hs = hist[b];
    __syncthreads();

    // per-query-row predicates (branchless ints)
    int rowok[8], segq[8];
#pragma unroll
    for (int r = 0; r < 8; r++) {
        const int qm = r + 8 * (lane >> 4);
        const int tq = qt * 16 + qm;
        rowok[r] = ((tq < vl) ? 1 : 0) | ((tq < hs) ? 2 : 0);
        segq[r] = sq[qm];
    }

    // each wave handles 4 key tiles of 16
    for (int i = 0; i < 4; i++) {
        const int kt = wave * 4 + i;
        const int tk = kt * 16 + (lane & 15);
        const int segk = seg[b * SEQ + tk];
        const int kok = (tk < vl) ? 1 : 0;
        const int khist = (tk < hs) ? 1 : 0;

        v8f acc;
#pragma unroll
        for (int j = 0; j < 8; j++) acc[j] = 0.0f;

        const unsigned short* krow =
            kb + (size_t)(b * SEQ + tk) * HDIM + h * DHEAD + (lane >> 4) * 16;
        Frag b0, b1;
        b0.q[0] = *(const uint4*)krow;        b0.q[1] = *(const uint4*)(krow + 8);
        b1.q[0] = *(const uint4*)(krow + 32); b1.q[1] = *(const uint4*)(krow + 40);

        acc = __builtin_amdgcn_wmma_f32_16x16x32_bf16(false, a0.v, false, b0.v, (short)0, acc, false, false);
        acc = __builtin_amdgcn_wmma_f32_16x16x32_bf16(false, a1.v, false, b1.v, (short)0, acc, false, false);

#pragma unroll
        for (int r = 0; r < 8; r++) {
            const int qm = r + 8 * (lane >> 4);
            const int ok = (rowok[r] & 1) & kok &
                           (((rowok[r] >> 1) & 1) | khist | ((segq[r] == segk) ? 1 : 0));
            s[qm][kt * 16 + (lane & 15)] = acc[r] * 0.125f + (ok ? 0.0f : -10000.0f);
        }
    }
    __syncthreads();

    // softmax over 512 keys: 16 rows x 16 groups of 32 cols
    const int r = tid >> 4, g = tid & 15;
    float mx = -3.0e38f;
    for (int c = g * 32; c < g * 32 + 32; c++) mx = fmaxf(mx, s[r][c]);
    red[r][g] = mx; __syncthreads();
    if (g == 0) {
        float m = red[r][0];
        for (int j = 1; j < 16; j++) m = fmaxf(m, red[r][j]);
        red[r][0] = m;
    }
    __syncthreads();
    const float rmax = red[r][0];
    __syncthreads();
    float sm = 0.0f;
    for (int c = g * 32; c < g * 32 + 32; c++) sm += expf(s[r][c] - rmax);
    red[r][g] = sm; __syncthreads();
    if (g == 0) {
        float t = 0.0f;
        for (int j = 0; j < 16; j++) t += red[r][j];
        red[r][0] = t;
    }
    __syncthreads();
    const float inv = 1.0f / red[r][0];

    unsigned short* prow =
        probs + (size_t)((b * NHEAD + h) * SEQ + qt * 16 + r) * SEQ;
    for (int c = g * 32; c < g * 32 + 32; c += 2) {
        unsigned lo = f2bf(expf(s[r][c] - rmax) * inv);
        unsigned hi = f2bf(expf(s[r][c + 1] - rmax) * inv);
        *(unsigned*)(prow + c) = lo | (hi << 16);
    }
}

// ---------------------------------------------------------------------------
// ctx = probs @ V  via WMMA (A = probs rows, B = V^T rows). 1 wave per block,
// one (b, h, 16-query tile) each; 64 WMMAs per block.
// ---------------------------------------------------------------------------
__global__ __launch_bounds__(32) void attn_ctx_kernel(
    const unsigned short* __restrict__ probs,
    const unsigned short* __restrict__ vt,
    unsigned short* __restrict__ ctxb)
{
    const int qt = blockIdx.x & 31;
    const int h  = (blockIdx.x >> 5) % NHEAD;
    const int b  = blockIdx.x / (32 * NHEAD);
    const int lane = threadIdx.x;

    v8f acc[4];
#pragma unroll
    for (int i = 0; i < 4; i++)
#pragma unroll
        for (int j = 0; j < 8; j++) acc[i][j] = 0.0f;

    const int ks = (lane & 16) ? 16 : 0;
    const unsigned short* prow =
        probs + (size_t)((b * NHEAD + h) * SEQ + qt * 16 + (lane & 15)) * SEQ;

    for (int k0 = 0; k0 < SEQ; k0 += 32) {
        Frag a;
        a.q[0] = *(const uint4*)(prow + k0 + ks);
        a.q[1] = *(const uint4*)(prow + k0 + ks + 8);
#pragma unroll
        for (int nt = 0; nt < 4; nt++) {
            const unsigned short* vrow =
                vt + (size_t)((b * NHEAD + h) * DHEAD + nt * 16 + (lane & 15)) * SEQ
                   + k0 + (lane >> 4) * 16;
            Frag bf;
            bf.q[0] = *(const uint4*)vrow;
            bf.q[1] = *(const uint4*)(vrow + 8);
            acc[nt] = __builtin_amdgcn_wmma_f32_16x16x32_bf16(
                false, a.v, false, bf.v, (short)0, acc[nt], false, false);
        }
    }
#pragma unroll
    for (int nt = 0; nt < 4; nt++)
#pragma unroll
        for (int r = 0; r < 8; r++) {
            const int tq = qt * 16 + r + 8 * (lane >> 4);
            ctxb[(size_t)(b * SEQ + tq) * HDIM + h * DHEAD + nt * 16 + (lane & 15)] =
                (unsigned short)f2bf(acc[nt][r]);
        }
}

// ---------------------------------------------------------------------------
// segment-mean pooling + linear head: out[b][s]
// ---------------------------------------------------------------------------
__global__ __launch_bounds__(256) void pool_logits_kernel(
    const float* __restrict__ x, const float* __restrict__ fcw,
    const float* __restrict__ fcb, const int* __restrict__ bos,
    const int* __restrict__ vlen, float* __restrict__ out)
{
    const int b = blockIdx.x / NSEG, sidx = blockIdx.x % NSEG;
    const int start = bos[b * NSEG + sidx];
    const int end = (sidx < NSEG - 1) ? bos[b * NSEG + sidx + 1] : vlen[b];
    __shared__ float red[256];
    float acc = 0.0f;
    for (int t = start; t < end; t++) {
        const float* xr = x + (size_t)(b * SEQ + t) * HDIM;
#pragma unroll
        for (int j = 0; j < 3; j++) {
            int d = threadIdx.x + j * 256;
            acc += xr[d] * fcw[d];
        }
    }
    red[threadIdx.x] = acc; __syncthreads();
    for (int off = 128; off > 0; off >>= 1) {
        if (threadIdx.x < off) red[threadIdx.x] += red[threadIdx.x + off];
        __syncthreads();
    }
    if (threadIdx.x == 0)
        out[b * NSEG + sidx] = red[0] / (float)(end - start) + fcb[0];
}

// ---------------------------------------------------------------------------
// host launcher
// ---------------------------------------------------------------------------
static inline size_t align256(size_t x) { return (x + 255) & ~(size_t)255; }

extern "C" void kernel_launch(void* const* d_in, const int* in_sizes, int n_in,
                              void* d_out, int out_size, void* d_ws, size_t ws_size,
                              hipStream_t stream)
{
    const float* word_emb = (const float*)d_in[0];
    const float* pos_emb  = (const float*)d_in[1];
    const float* type_emb = (const float*)d_in[2];
    const float* emb_ln_g = (const float*)d_in[3];
    const float* emb_ln_b = (const float*)d_in[4];
    const float* Wq = (const float*)d_in[5];
    const float* bq = (const float*)d_in[6];
    const float* Wk = (const float*)d_in[7];
    const float* bk = (const float*)d_in[8];
    const float* Wv = (const float*)d_in[9];
    const float* bv = (const float*)d_in[10];
    const float* Wo = (const float*)d_in[11];
    const float* bo = (const float*)d_in[12];
    const float* attn_ln_g = (const float*)d_in[13];
    const float* attn_ln_b = (const float*)d_in[14];
    const float* W1 = (const float*)d_in[15];
    const float* b1 = (const float*)d_in[16];
    const float* W2 = (const float*)d_in[17];
    const float* b2 = (const float*)d_in[18];
    const float* ffn_ln_g = (const float*)d_in[19];
    const float* ffn_ln_b = (const float*)d_in[20];
    const float* fc_w = (const float*)d_in[21];
    const float* fc_b = (const float*)d_in[22];
    const int* input_ids = (const int*)d_in[23];
    const int* token_type = (const int*)d_in[24];
    const int* attn_mask = (const int*)d_in[25];
    const int* bos = (const int*)d_in[26];

    // workspace carve-up (~69 MB)
    char* p = (char*)d_ws;
    size_t off = 0;
    float* x = (float*)(p + off);                 off = align256(off + (size_t)NROW * HDIM * 4);
    float* tmp = (float*)(p + off);               off = align256(off + (size_t)NROW * HDIM * 4);
    unsigned short* xb   = (unsigned short*)(p + off); off = align256(off + (size_t)NROW * HDIM * 2);
    unsigned short* qb   = (unsigned short*)(p + off); off = align256(off + (size_t)NROW * HDIM * 2);
    unsigned short* kb   = (unsigned short*)(p + off); off = align256(off + (size_t)NROW * HDIM * 2);
    unsigned short* vb   = (unsigned short*)(p + off); off = align256(off + (size_t)NROW * HDIM * 2);
    unsigned short* vt   = (unsigned short*)(p + off); off = align256(off + (size_t)NROW * HDIM * 2);
    unsigned short* ctxb = (unsigned short*)(p + off); off = align256(off + (size_t)NROW * HDIM * 2);
    unsigned short* hb   = (unsigned short*)(p + off); off = align256(off + (size_t)NROW * FFDIM * 2);
    unsigned short* probs = (unsigned short*)(p + off);
    off = align256(off + (size_t)BATCH * NHEAD * SEQ * SEQ * 2);
    int* seg  = (int*)(p + off); off = align256(off + (size_t)NROW * 4);
    int* vlen = (int*)(p + off); off = align256(off + (size_t)BATCH * 4);
    int* hist = (int*)(p + off); off = align256(off + (size_t)BATCH * 4);
    (void)ws_size; (void)n_in; (void)in_sizes; (void)out_size;

    setup_mask_kernel<<<8, 256, 0, stream>>>(attn_mask, bos, seg, vlen, hist);
    embed_ln_kernel<<<NROW, 256, 0, stream>>>(word_emb, pos_emb, type_emb,
                                              emb_ln_g, emb_ln_b,
                                              input_ids, token_type, x, xb);

    const dim3 blk(256);
    const dim3 grid_h(HDIM / BN, NROW / BM);     // (6,16)
    const dim3 grid_ff(FFDIM / BN, NROW / BM);   // (24,16)
    const int attn_blocks = BATCH * NHEAD * (SEQ / 16); // 1536

    for (int l = 0; l < NLAYER; l++) {
        const size_t wo_h = (size_t)l * HDIM * HDIM;
        const size_t wo_f = (size_t)l * HDIM * FFDIM;
        const size_t bo_h = (size_t)l * HDIM;
        const size_t bo_f = (size_t)l * FFDIM;

        gemm_bf16_kernel<1><<<grid_h, blk, 0, stream>>>(xb, Wq + wo_h, bq + bo_h, qb, NROW, HDIM, HDIM);
        gemm_bf16_kernel<1><<<grid_h, blk, 0, stream>>>(xb, Wk + wo_h, bk + bo_h, kb, NROW, HDIM, HDIM);
        gemm_bf16_kernel<1><<<grid_h, blk, 0, stream>>>(xb, Wv + wo_h, bv + bo_h, vb, NROW, HDIM, HDIM);
        transpose_v_kernel<<<(NROW * HDIM + 255) / 256, 256, 0, stream>>>(vb, vt);
        attn_scores_softmax_kernel<<<attn_blocks, 256, 0, stream>>>(qb, kb, probs, seg, vlen, hist);
        attn_ctx_kernel<<<attn_blocks, 32, 0, stream>>>(probs, vt, ctxb);
        gemm_bf16_kernel<0><<<grid_h, blk, 0, stream>>>(ctxb, Wo + wo_h, bo + bo_h, tmp, NROW, HDIM, HDIM);
        add_ln_kernel<<<NROW, 256, 0, stream>>>(x, tmp, attn_ln_g + bo_h, attn_ln_b + bo_h, xb);
        gemm_bf16_kernel<2><<<grid_ff, blk, 0, stream>>>(xb, W1 + wo_f, b1 + bo_f, hb, NROW, FFDIM, HDIM);
        gemm_bf16_kernel<0><<<grid_h, blk, 0, stream>>>(hb, W2 + wo_f, b2 + bo_h, tmp, NROW, HDIM, FFDIM);
        add_ln_kernel<<<NROW, 256, 0, stream>>>(x, tmp, ffn_ln_g + bo_h, ffn_ln_b + bo_h, xb);
    }

    pool_logits_kernel<<<BATCH * NSEG, 256, 0, stream>>>(x, fc_w, fc_b, bos, vlen, (float*)d_out);
}
